// UnsupervisedGCN_86431921864946
// MI455X (gfx1250) — matrix operand order repaired and checked by reading, not compiled, not run on hardware
//
#include <hip/hip_runtime.h>
#include <hip/hip_bf16.h>
#include <math.h>

typedef __attribute__((ext_vector_type(16))) _Float16 v16h;
typedef __attribute__((ext_vector_type(8)))  _Float16 v8h;
typedef __attribute__((ext_vector_type(8)))  float    v8f;

// ---------------------------------------------------------------------------
// Degree / normalization
// ---------------------------------------------------------------------------
__global__ void gcn_deg_init(float* deg, int n) {
    int i = blockIdx.x * blockDim.x + threadIdx.x;
    if (i < n) deg[i] = 1.0f;  // self-loop contributes 1 to every node's degree
}

__global__ void gcn_deg_count(const int* __restrict__ col, float* deg, int e) {
    int i = blockIdx.x * blockDim.x + threadIdx.x;
    if (i < e) atomicAdd(&deg[col[i]], 1.0f);
}

__global__ void gcn_make_dinv(float* deg, int n) {
    int i = blockIdx.x * blockDim.x + threadIdx.x;
    if (i < n) deg[i] = rsqrtf(deg[i]);  // deg >= 1 always (self loops)
}

// ---------------------------------------------------------------------------
// Activations: fp32 -> fp16, K padded to Kp (zero-fill k >= fi), row-major
// ---------------------------------------------------------------------------
__global__ void gcn_cvt_f16_pad(const float* __restrict__ src, _Float16* __restrict__ dst,
                                int rows, int fi, int Kp) {
    int idx = blockIdx.x * blockDim.x + threadIdx.x;
    int total = rows * Kp;
    if (idx >= total) return;
    int r = idx / Kp;
    int k = idx - r * Kp;
    dst[idx] = (k < fi) ? (_Float16)src[(size_t)r * fi + k] : (_Float16)0.0f;
}

// ---------------------------------------------------------------------------
// Weights: fp32 [fi x fo] -> fp16 pre-swizzled into WMMA B-fragment order.
// Bs[((kTile*NT + nTile)*32 + lane)*16 + t] = W[k][n], with
//   k = kTile*32 + (lane>=16 ? 8 : 0) + (t&7) + (t>=8 ? 16 : 0),  n = nTile*16 + (lane&15)
// so the GEMM reads each lane's B fragment as 32 contiguous bytes.
// ---------------------------------------------------------------------------
__global__ void gcn_w_swizzle(const float* __restrict__ W, _Float16* __restrict__ Bs,
                              int fi, int fo, int Kp) {
    int idx = blockIdx.x * blockDim.x + threadIdx.x;
    int total = Kp * fo;
    if (idx >= total) return;
    int t      = idx & 15;
    int lane   = (idx >> 4) & 31;
    int rest   = idx >> 9;            // kTile*NT + nTile   (32*16 = 512 elems per pair)
    int ntiles = fo >> 4;
    int nTile  = rest % ntiles;
    int kTile  = rest / ntiles;
    int k = kTile * 32 + ((lane >> 4) << 3) + (t & 7) + ((t >> 3) << 4);
    int n = nTile * 16 + (lane & 15);
    Bs[idx] = (k < fi) ? (_Float16)W[(size_t)k * fo + n] : (_Float16)0.0f;
}

// ---------------------------------------------------------------------------
// WMMA GEMM: C[N x FO] = A16[N x K] * Bs (pre-swizzled), K multiple of 32.
// One wave owns a 16-row strip and ALL FO/16 column tiles: A fragment loaded
// once per k-step (2 x b128), then FO/16 back-to-back v_wmma ops.
// ---------------------------------------------------------------------------
template <int FO>
__global__ __launch_bounds__(256) void gcn_gemm_wmma(
    const _Float16* __restrict__ A, const _Float16* __restrict__ Bs,
    float* __restrict__ C, int Nrows, int K) {
    constexpr int NT = FO / 16;
    const int wave = threadIdx.x >> 5;
    const int lane = threadIdx.x & 31;
    const int mTile = blockIdx.x * 8 + wave;
    if (mTile * 16 >= Nrows) return;          // wave-uniform: EXEC stays all-ones
    const int m0 = mTile * 16;

    // A fragment (16-bit A 16x32): lane L<16 -> row M=L,    K = 0..7  and 16..23
    //                              lane L>=16 -> row M=L-16, K = 8..15 and 24..31
    const int am  = lane & 15;
    const int akb = (lane >> 4) * 8;
    const _Float16* __restrict__ arow = A + (size_t)(m0 + am) * K + akb;

    v8f acc[NT];
#pragma unroll
    for (int nt = 0; nt < NT; ++nt) acc[nt] = (v8f){};

    const int kTiles = K >> 5;
    for (int kt = 0; kt < kTiles; ++kt) {
        v8h alo = *(const v8h*)(arow + kt * 32);        // 16B -> global_load_b128
        v8h ahi = *(const v8h*)(arow + kt * 32 + 16);   // 16B -> global_load_b128
        v16h a = __builtin_shufflevector(alo, ahi,
                                         0, 1, 2, 3, 4, 5, 6, 7,
                                         8, 9, 10, 11, 12, 13, 14, 15);
        const _Float16* __restrict__ bp =
            Bs + ((size_t)(kt * NT) * 32 + lane) * 16;
#pragma unroll
        for (int nt = 0; nt < NT; ++nt) {
            v16h b = *(const v16h*)(bp + (size_t)nt * 32 * 16);  // 32B contiguous
            acc[nt] = __builtin_amdgcn_wmma_f32_16x16x32_f16(
                /*neg_a=*/false, a, /*neg_b=*/false, b,
                /*c_mod=*/(short)0, acc[nt], /*reuse_a=*/false, /*reuse_b=*/false);
        }
    }

    // C/D layout: VGPR r -> row (lane>>4)*8 + r, col lane&15 (coalesced b32 stores)
    const int cm = m0 + (lane >> 4) * 8;
    const int cn = lane & 15;
#pragma unroll
    for (int nt = 0; nt < NT; ++nt) {
#pragma unroll
        for (int r = 0; r < 8; ++r) {
            C[(size_t)(cm + r) * FO + nt * 16 + cn] = acc[nt][r];
        }
    }
}

// ---------------------------------------------------------------------------
// agg[i,f] = hW[i,f] * dinv[i]^2  (self-loop term; fully initializes agg)
// ---------------------------------------------------------------------------
__global__ void gcn_self_init(const float* __restrict__ hW, const float* __restrict__ dinv,
                              float* __restrict__ agg, int n, int fo) {
    int idx = blockIdx.x * blockDim.x + threadIdx.x;
    if (idx >= n * fo) return;
    int i = idx / fo;
    float d = dinv[i];
    agg[idx] = hW[idx] * d * d;
}

// ---------------------------------------------------------------------------
// Edge scatter: agg[col[e], f] += hW[row[e], f] * dinv[row]*dinv[col]
// threadIdx.x = feature lane (coalesced gather + coalesced atomics, L2-resident)
// ---------------------------------------------------------------------------
__global__ void gcn_scatter(const float* __restrict__ hW,
                            const int* __restrict__ row, const int* __restrict__ col,
                            const float* __restrict__ dinv, float* __restrict__ agg,
                            int E, int fo) {
    int f = threadIdx.x;
    int e = blockIdx.x * blockDim.y + threadIdx.y;
    if (e >= E) return;
    int r = row[e];
    int c = col[e];
    float w = dinv[r] * dinv[c];
    atomicAdd(&agg[(size_t)c * fo + f], hW[(size_t)r * fo + f] * w);
}

// ---------------------------------------------------------------------------
// dst[i,f] = (relu?) (src[i,f] + b[f])
// ---------------------------------------------------------------------------
__global__ void gcn_bias_act(const float* __restrict__ src, const float* __restrict__ b,
                             float* __restrict__ dst, int n, int fo, int do_relu) {
    int idx = blockIdx.x * blockDim.x + threadIdx.x;
    if (idx >= n * fo) return;
    int f = idx % fo;
    float v = src[idx] + b[f];
    dst[idx] = do_relu ? fmaxf(v, 0.0f) : v;
}

// ---------------------------------------------------------------------------
// Host orchestration
// ---------------------------------------------------------------------------
static inline size_t align256(size_t x) { return (x + 255) & ~(size_t)255; }

static void launch_gemm(int fo, const _Float16* A, const _Float16* Bs, float* C,
                        int N, int K, hipStream_t stream) {
    dim3 grid((N / 16 + 7) / 8);
    switch (fo) {
        case 16:  gcn_gemm_wmma<16> <<<grid, 256, 0, stream>>>(A, Bs, C, N, K); break;
        case 32:  gcn_gemm_wmma<32> <<<grid, 256, 0, stream>>>(A, Bs, C, N, K); break;
        case 64:  gcn_gemm_wmma<64> <<<grid, 256, 0, stream>>>(A, Bs, C, N, K); break;
        default:  gcn_gemm_wmma<128><<<grid, 256, 0, stream>>>(A, Bs, C, N, K); break;
    }
}

extern "C" void kernel_launch(void* const* d_in, const int* in_sizes, int n_in,
                              void* d_out, int out_size, void* d_ws, size_t ws_size,
                              hipStream_t stream) {
    const int F_IN = 128;
    const int N = in_sizes[0] / F_IN;          // 100000
    const int E = in_sizes[1] / 2;             // 3200000

    const float* x   = (const float*)d_in[0];
    const int*   ei  = (const int*)d_in[1];
    const int*   row = ei;                     // edge_index[0] = source j
    const int*   col = ei + E;                 // edge_index[1] = target i

    const float* W[7];
    const float* Bv[7];
    for (int l = 0; l < 6; ++l) { W[l] = (const float*)d_in[3 + 2 * l]; Bv[l] = (const float*)d_in[4 + 2 * l]; }
    W[6]  = (const float*)d_in[15];            // Wf
    Bv[6] = (const float*)d_in[16];            // bf

    static const int FI[7] = {128, 64, 32, 16, 32, 64, 128};
    static const int FO[7] = { 64, 32, 16, 32, 64, 128, 128};

    // Workspace carve-up (~180 MB)
    char* ws = (char*)d_ws;
    size_t off = 0;
    float*    dinv = (float*)(ws + off);    off += align256((size_t)N * 4);
    float*    h    = (float*)(ws + off);    off += align256((size_t)N * F_IN * 4);
    float*    hW   = (float*)(ws + off);    off += align256((size_t)N * F_IN * 4);
    float*    agg  = (float*)(ws + off);    off += align256((size_t)N * F_IN * 4);
    _Float16* h16  = (_Float16*)(ws + off); off += align256((size_t)N * F_IN * 2);
    _Float16* w16  = (_Float16*)(ws + off); off += align256((size_t)F_IN * F_IN * 2);
    (void)ws_size;

    const int TPB = 256;

    // 1) degrees -> dinv
    gcn_deg_init<<<(N + TPB - 1) / TPB, TPB, 0, stream>>>(dinv, N);
    gcn_deg_count<<<(E + TPB - 1) / TPB, TPB, 0, stream>>>(col, dinv, E);
    gcn_make_dinv<<<(N + TPB - 1) / TPB, TPB, 0, stream>>>(dinv, N);

    // 2) six GCN conv layers + final dense
    for (int l = 0; l < 7; ++l) {
        const int fi = FI[l], fo = FO[l];
        const int Kp = (fi + 31) & ~31;
        const float* h_src = (l == 0) ? x : h;

        // activations fp32 -> fp16 (K padded)
        {
            int total = N * Kp;
            gcn_cvt_f16_pad<<<(total + TPB - 1) / TPB, TPB, 0, stream>>>(h_src, h16, N, fi, Kp);
        }
        // weights fp32 -> fp16, pre-swizzled into B-fragment order (K padded)
        {
            int total = Kp * fo;
            gcn_w_swizzle<<<(total + TPB - 1) / TPB, TPB, 0, stream>>>(W[l], w16, fi, fo, Kp);
        }

        // WMMA GEMM: hW = h16 @ W
        launch_gemm(fo, h16, w16, hW, N, Kp, stream);

        if (l < 6) {
            int total = N * fo;
            gcn_self_init<<<(total + TPB - 1) / TPB, TPB, 0, stream>>>(hW, dinv, agg, N, fo);

            dim3 blk(fo, 256 / fo);
            dim3 grd((E + blk.y - 1) / blk.y);
            gcn_scatter<<<grd, blk, 0, stream>>>(hW, row, col, dinv, agg, E, fo);

            gcn_bias_act<<<(total + TPB - 1) / TPB, TPB, 0, stream>>>(agg, Bv[l], h, N, fo, 1);
        } else {
            int total = N * fo;
            gcn_bias_act<<<(total + TPB - 1) / TPB, TPB, 0, stream>>>(hW, Bv[l], (float*)d_out, N, fo, 0);
        }
    }
}
